// HeReGAT_nc_FC_5909875000129
// MI455X (gfx1250) — matrix operand order repaired and checked by reading, not compiled, not run on hardware
//
#include <hip/hip_runtime.h>
#include <hip/hip_bf16.h>
#include <math.h>

// ---------------- CDNA5 WMMA types ----------------
typedef __attribute__((ext_vector_type(16))) __bf16 v16bf;
typedef __attribute__((ext_vector_type(8)))  float  v8f;
typedef int v4i_async __attribute__((vector_size(16)));   // arg type for async-to-LDS builtin

// ---------------- problem constants ----------------
#define NT     9000
#define HDIM   128
#define NHEADS 4
#define NCLS   3

static const int g_counts[3] = {4000, 3000, 2000};
static const int g_starts[3] = {0, 4000, 7000};

// ---------------- GEMM tile config ----------------
// 128x128 block tile, 8 waves (2 row-groups x 4 col-groups), 64x32 per wave,
// 4x2 WMMA accumulators per wave -> 8 v_wmma per 32-deep k-step.
// LDS row stride 40 bf16 (80B): keeps 16B alignment for async B128 writes.
enum { TBM = 128, TBN = 128, TBK = 32, LSTR = 40 };

#if defined(__HIP_DEVICE_COMPILE__) && __has_builtin(__builtin_amdgcn_global_load_async_to_lds_b128)
#define ASYNC_OK 1
#else
#define ASYNC_OK 0
#endif

__device__ __forceinline__ unsigned short f2bf_bits(float x) {
  unsigned u = __float_as_uint(x);
  return (unsigned short)((u + 0x7fffu + ((u >> 16) & 1u)) >> 16);
}

union Frag { unsigned u[8]; v16bf v; };

// C[M,N] = act(alpha * A @ B^T_prep + bias (+C_old))
// A: bf16 [M][K] row-major (lda==K). B: bf16 [N][K] row-major (ldb==K).
__global__ __launch_bounds__(256)
void gemm_wmma(const unsigned short* __restrict__ A, int lda,
               const unsigned short* __restrict__ B, int ldb,
               float* __restrict__ C, int ldc,
               const float* __restrict__ bias,
               int M, int N, int K,
               int accumulate, int act, float alpha) {
  __shared__ unsigned short sA[TBM][LSTR];
  __shared__ unsigned short sB[TBN][LSTR];

  const int tid  = threadIdx.x;
  const int lane = tid & 31;
  const int w    = tid >> 5;          // 8 waves
  const int wrow = w & 1;             // 2 wave-rows (64 rows each)
  const int wcol = w >> 1;            // 4 wave-cols (32 cols each)
  const int half = lane >> 4;
  const int l16  = lane & 15;
  const int bm0  = blockIdx.y * TBM;
  const int bn0  = blockIdx.x * TBN;

  const bool intA = (bm0 + TBM <= M);
  const bool intB = (bn0 + TBN <= N);
  const bool kAligned = ((K & 7) == 0);

  v8f acc[4][2] = {};

  for (int k0 = 0; k0 < K; k0 += TBK) {
    const bool fullK = (k0 + TBK <= K);
    if (fullK && k0 + TBK < K) {  // prefetch next tiles -> global_prefetch_b8
      __builtin_prefetch(A + (size_t)bm0 * lda + (k0 + TBK), 0, 1);
      __builtin_prefetch(B + (size_t)bn0 * ldb + (k0 + TBK), 0, 1);
    }
#if ASYNC_OK
    if (intA && fullK && kAligned) {
      // 128 rows x 4 x 16B segments = 512 async B128 copies, 2/thread
      #pragma unroll
      for (int i = 0; i < 2; ++i) {
        int idx = i * 256 + tid;
        int r = idx >> 2, seg = idx & 3;
        __builtin_amdgcn_global_load_async_to_lds_b128(
            (v4i_async*)(A + (size_t)(bm0 + r) * lda + k0 + seg * 8),
            (v4i_async*)(&sA[r][seg * 8]), 0, 0);
      }
    } else
#endif
    {
      #pragma unroll
      for (int i = 0; i < 16; ++i) {
        int idx = i * 256 + tid;
        int r = idx >> 5, c = idx & 31;
        int gr = bm0 + r, gc = k0 + c;
        sA[r][c] = (gr < M && gc < K) ? A[(size_t)gr * lda + gc] : (unsigned short)0;
      }
    }
#if ASYNC_OK
    if (intB && fullK && kAligned) {
      #pragma unroll
      for (int i = 0; i < 2; ++i) {
        int idx = i * 256 + tid;
        int r = idx >> 2, seg = idx & 3;
        __builtin_amdgcn_global_load_async_to_lds_b128(
            (v4i_async*)(B + (size_t)(bn0 + r) * ldb + k0 + seg * 8),
            (v4i_async*)(&sB[r][seg * 8]), 0, 0);
      }
    } else
#endif
    {
      #pragma unroll
      for (int i = 0; i < 16; ++i) {
        int idx = i * 256 + tid;
        int r = idx >> 5, c = idx & 31;
        int gn = bn0 + r, gc = k0 + c;
        sB[r][c] = (gn < N && gc < K) ? B[(size_t)gn * ldb + gc] : (unsigned short)0;
      }
    }
#if ASYNC_OK
#if __has_builtin(__builtin_amdgcn_s_wait_asynccnt)
    __builtin_amdgcn_s_wait_asynccnt(0);
#else
    asm volatile("s_wait_asynccnt 0" ::: "memory");
#endif
#endif
    __syncthreads();

    // A fragment: lane holds row l16; K = {half*8+2q}q<4  U  {16+half*8+2(q-4)}
    Frag fa[4], fb[2];
    #pragma unroll
    for (int sm = 0; sm < 4; ++sm) {
      int row = wrow * 64 + sm * 16 + l16;
      #pragma unroll
      for (int q = 0; q < 8; ++q) {
        int k = (q < 4) ? (half * 8 + 2 * q) : (16 + half * 8 + 2 * (q - 4));
        fa[sm].u[q] = *reinterpret_cast<const unsigned*>(&sA[row][k]);
      }
    }
    // B fragment: lane holds col l16; K = {half*8+2q}q<4  U  {16+half*8+2(q-4)}
    #pragma unroll
    for (int sn = 0; sn < 2; ++sn) {
      int col = wcol * 32 + sn * 16 + l16;
      #pragma unroll
      for (int q = 0; q < 8; ++q) {
        int k = (q < 4) ? (half * 8 + 2 * q) : (16 + half * 8 + 2 * (q - 4));
        fb[sn].u[q] = *reinterpret_cast<const unsigned*>(&sB[col][k]);
      }
    }
    #pragma unroll
    for (int sm = 0; sm < 4; ++sm)
      #pragma unroll
      for (int sn = 0; sn < 2; ++sn)
        acc[sm][sn] = __builtin_amdgcn_wmma_f32_16x16x32_bf16(
            false, fa[sm].v, false, fb[sn].v, (short)0, acc[sm][sn], false, false);
    __syncthreads();
  }

  // epilogue: VGPR r -> (row r + 8*half, col l16)
  #pragma unroll
  for (int sm = 0; sm < 4; ++sm)
    #pragma unroll
    for (int sn = 0; sn < 2; ++sn)
      #pragma unroll
      for (int r = 0; r < 8; ++r) {
        int row = bm0 + wrow * 64 + sm * 16 + r + half * 8;
        int col = bn0 + wcol * 32 + sn * 16 + l16;
        if (row < M && col < N) {
          float vv = acc[sm][sn][r] * alpha;
          if (bias) vv += bias[col];
          if (accumulate) vv += C[(size_t)row * ldc + col];
          if (act == 1) vv = vv > 0.f ? vv : (expf(vv) - 1.f);  // ELU
          C[(size_t)row * ldc + col] = vv;
        }
      }
}

// fp32 -> bf16 operand prep; transpose=1: src R x C row-major -> dst C x R
__global__ void cvt_bf16(const float* __restrict__ src, unsigned short* __restrict__ dst,
                         int R, int Cc, int transpose) {
  size_t i = (size_t)blockIdx.x * 256 + threadIdx.x;
  if (i >= (size_t)R * Cc) return;
  int r = (int)(i / Cc), c = (int)(i - (size_t)r * Cc);
  unsigned short h = f2bf_bits(src[i]);
  if (transpose) dst[(size_t)c * R + r] = h;
  else           dst[i] = h;
}

// ---------------- small utility kernels ----------------
__global__ void fill_f32(float* p, float v, int n) {
  int i = blockIdx.x * 256 + threadIdx.x;
  if (i < n) p[i] = v;
}

__device__ __forceinline__ void atomicMaxF(float* addr, float v) {
  int* ai = reinterpret_cast<int*>(addr);
  int old = *ai;
  while (__int_as_float(old) < v) {
    int assumed = old;
    old = atomicCAS(ai, assumed, __float_as_int(v));
    if (old == assumed) break;
  }
}

__global__ void attn_dots(const float* __restrict__ feat,
                          const float* __restrict__ al, const float* __restrict__ ar,
                          float* __restrict__ el, float* __restrict__ er, int heads) {
  int node = blockIdx.x;
  int lane = threadIdx.x & 31, h = threadIdx.x >> 5;
  if (h >= heads) return;
  const float* f = feat + (size_t)node * heads * HDIM + h * HDIM;
  float sl = 0.f, sr = 0.f;
  for (int d = lane; d < HDIM; d += 32) {
    float x = f[d];
    sl += x * al[h * HDIM + d];
    sr += x * ar[h * HDIM + d];
  }
  for (int o = 16; o; o >>= 1) { sl += __shfl_xor(sl, o); sr += __shfl_xor(sr, o); }
  if (lane == 0) { el[node * heads + h] = sl; er[node * heads + h] = sr; }
}

__global__ void edge_logit_max(const int* __restrict__ src, const int* __restrict__ dst,
                               const float* __restrict__ el, const float* __restrict__ er,
                               float* __restrict__ ebuf, float* __restrict__ mx,
                               int E, int heads) {
  int i = blockIdx.x * 256 + threadIdx.x;
  if (i >= E * heads) return;
  int e = i / heads, h = i - e * heads;
  float v = el[src[e] * heads + h] + er[dst[e] * heads + h];
  v = v > 0.f ? v : 0.2f * v;   // leaky relu
  ebuf[i] = v;
  atomicMaxF(&mx[dst[e] * heads + h], v);
}

__global__ void edge_expsum(const int* __restrict__ dst, const float* __restrict__ mx,
                            float* __restrict__ ebuf, float* __restrict__ den,
                            int E, int heads) {
  int i = blockIdx.x * 256 + threadIdx.x;
  if (i >= E * heads) return;
  int e = i / heads, h = i - e * heads;
  float ex = expf(ebuf[i] - mx[dst[e] * heads + h]);
  ebuf[i] = ex;
  atomicAdd(&den[dst[e] * heads + h], ex);
}

__global__ void edge_aggr(const int* __restrict__ src, const int* __restrict__ dst,
                          const float* __restrict__ ebuf, const float* __restrict__ den,
                          const float* __restrict__ feat, float* __restrict__ out,
                          int heads) {
  int e = blockIdx.x;
  int t = threadIdx.x, h = t >> 5, lane = t & 31;
  if (h >= heads) return;
  int s = src[e], d = dst[e];
  float a = ebuf[(size_t)e * heads + h] / den[d * heads + h];
  const float* fs = feat + (size_t)s * heads * HDIM + h * HDIM;
  float* od = out + (size_t)d * heads * HDIM + h * HDIM;
  for (int j = lane; j < HDIM; j += 32) atomicAdd(&od[j], a * fs[j]);
}

__global__ void ew_finish(float* __restrict__ out, const float* __restrict__ res,
                          const float* __restrict__ bias, int n, int cols, int act) {
  int i = blockIdx.x * 256 + threadIdx.x;
  if (i >= n * cols) return;
  int j = i % cols;
  float v = out[i];
  if (res) v += res[i];
  v += bias[j];
  if (act) v = v > 0.f ? v : (expf(v) - 1.f);
  out[i] = v;
}

__global__ void masked_softmax(float* __restrict__ logits, int ldl,
                               const float* __restrict__ adj, int ldadj,
                               int rowoff, int coloff, int ncols) {
  int m = blockIdx.x;
  float* row = logits + (size_t)m * ldl;
  const float* arow = adj + (size_t)(rowoff + m) * ldadj + coloff;
  __shared__ float red[256];
  int t = threadIdx.x;
  float mx = -1e30f;
  for (int n = t; n < ncols; n += 256) {
    float v = arow[n] > 0.f ? row[n] : -1e9f;
    row[n] = v;
    mx = fmaxf(mx, v);
  }
  red[t] = mx; __syncthreads();
  for (int s = 128; s; s >>= 1) { if (t < s) red[t] = fmaxf(red[t], red[t + s]); __syncthreads(); }
  mx = red[0]; __syncthreads();
  float sm = 0.f;
  for (int n = t; n < ncols; n += 256) { float ex = expf(row[n] - mx); row[n] = ex; sm += ex; }
  red[t] = sm; __syncthreads();
  for (int s = 128; s; s >>= 1) { if (t < s) red[t] += red[t + s]; __syncthreads(); }
  float inv = 1.f / red[0];
  for (int n = t; n < ncols; n += 256) row[n] *= inv;
}

// ---------------- host-side parameter unpacking ----------------
struct HGP {
  const float *fcW[3], *fcb[3];
  const float *al[3], *ar[3], *bias[3], *fcw[3], *resw[3];
  const float *linW, *linb;
  const float *ntW[3], *ntb[3];
};
struct Params {
  const float *fc_top_W[3], *fc_top_b[3];
  const float *hgn_W;
  HGP hg[2];
};

static void build_params(const float* P, Params& pp) {
  size_t o = 0;
  auto take = [&](size_t n) { const float* r = P + o; o += n; return r; };
  for (int i = 0; i < 3; ++i) { pp.fc_top_W[i] = take(128 * 128); pp.fc_top_b[i] = take(128); }
  pp.hgn_W = take((size_t)4 * 128 * 128);
  for (int L = 0; L < 2; ++L) {
    HGP& g = pp.hg[L];
    for (int i = 0; i < 3; ++i) {
      size_t in = L ? 128 : (size_t)g_counts[i];
      g.fcW[i] = take(in * 128); g.fcb[i] = take(128);
    }
    g.al[0] = take(4 * 128); g.ar[0] = take(4 * 128); g.bias[0] = take(512);
    g.fcw[0] = take((size_t)256 * 512); g.resw[0] = take((size_t)256 * 512);
    g.al[1] = take(4 * 128); g.ar[1] = take(4 * 128); g.bias[1] = take(512);
    g.fcw[1] = take((size_t)512 * 512); g.resw[1] = nullptr;
    g.al[2] = take(128); g.ar[2] = take(128); g.bias[2] = take(128);
    g.fcw[2] = take((size_t)512 * 128); g.resw[2] = take((size_t)512 * 128);
    g.linW = take(128 * 3); g.linb = take(3);
    for (int i = 0; i < 3; ++i) { g.ntW[i] = take(3 * 128); g.ntb[i] = take(128); }
  }
}

// ---------------- host launch helpers ----------------
static inline void fill(hipStream_t s, float* p, float v, int n) {
  fill_f32<<<(n + 255) / 256, 256, 0, s>>>(p, v, n);
}
static inline void cvt(hipStream_t s, const float* src, unsigned short* dst,
                       int R, int Cc, int tr) {
  size_t tot = (size_t)R * Cc;
  cvt_bf16<<<(unsigned)((tot + 255) / 256), 256, 0, s>>>(src, dst, R, Cc, tr);
}
// bKxN=1: B is K x N row-major (weights) -> prepped transposed. 0: already N x K.
static void gemm_auto(hipStream_t s, const float* A, const float* B, int bKxN,
                      float* C, int ldc, const float* bias, int M, int N, int K,
                      int accum, int act, float alpha,
                      unsigned short* bfA, unsigned short* bfB) {
  cvt(s, A, bfA, M, K, 0);
  if (bKxN) cvt(s, B, bfB, K, N, 1);
  else      cvt(s, B, bfB, N, K, 0);
  dim3 g((N + TBN - 1) / TBN, (M + TBM - 1) / TBM);
  gemm_wmma<<<g, 256, 0, s>>>(bfA, K, bfB, K, C, ldc, bias, M, N, K, accum, act, alpha);
}

struct WS {
  unsigned short *bfA, *bfB;
  float *hbuf, *featb, *Abuf, *Bbuf, *resb, *el, *er, *mx, *den, *ebuf, *h1c, *h2c, *attn;
};

static void run_gat_attention(hipStream_t s, const int* src, const int* dst, int E, int heads,
                              const float* feat, const HGP& g, int gi,
                              float* outbuf, const WS& w) {
  attn_dots<<<NT, heads * 32, 0, s>>>(feat, g.al[gi], g.ar[gi], w.el, w.er, heads);
  fill(s, w.mx, -1e30f, NT * heads);
  fill(s, w.den, 0.f, NT * heads);
  int tot = E * heads;
  edge_logit_max<<<(tot + 255) / 256, 256, 0, s>>>(src, dst, w.el, w.er, w.ebuf, w.mx, E, heads);
  edge_expsum<<<(tot + 255) / 256, 256, 0, s>>>(dst, w.mx, w.ebuf, w.den, E, heads);
  edge_aggr<<<E, heads * 32, 0, s>>>(src, dst, w.ebuf, w.den, feat, outbuf, heads);
}

static void run_hgat(hipStream_t s, const HGP& g,
                     const float* const feats[3], const int fdims[3],
                     const float* const ntfs[3],
                     const int* src, const int* dst, int E,
                     float* outH, const WS& w) {
  for (int i = 0; i < 3; ++i) {
    gemm_auto(s, feats[i], g.fcW[i], 1, w.hbuf + (size_t)g_starts[i] * 256, 256,
              g.fcb[i], g_counts[i], 128, fdims[i], 0, 0, 1.f, w.bfA, w.bfB);
    gemm_auto(s, ntfs[i], g.ntW[i], 1, w.hbuf + (size_t)g_starts[i] * 256 + 128, 256,
              g.ntb[i], g_counts[i], 128, 3, 0, 0, 1.f, w.bfA, w.bfB);
  }
  // GAT 0: (NT,256) -> (NT,512), no residual, ELU
  gemm_auto(s, w.hbuf, g.fcw[0], 1, w.featb, 512, nullptr, NT, 512, 256, 0, 0, 1.f, w.bfA, w.bfB);
  fill(s, w.Abuf, 0.f, NT * 512);
  run_gat_attention(s, src, dst, E, NHEADS, w.featb, g, 0, w.Abuf, w);
  ew_finish<<<(NT * 512 + 255) / 256, 256, 0, s>>>(w.Abuf, nullptr, g.bias[0], NT, 512, 1);
  // GAT 1: (NT,512) -> (NT,512), identity residual, ELU
  gemm_auto(s, w.Abuf, g.fcw[1], 1, w.featb, 512, nullptr, NT, 512, 512, 0, 0, 1.f, w.bfA, w.bfB);
  fill(s, w.Bbuf, 0.f, NT * 512);
  run_gat_attention(s, src, dst, E, NHEADS, w.featb, g, 1, w.Bbuf, w);
  ew_finish<<<(NT * 512 + 255) / 256, 256, 0, s>>>(w.Bbuf, w.Abuf, g.bias[1], NT, 512, 1);
  // final GAT: heads=1, (NT,512) -> (NT,128), projected residual, no act
  gemm_auto(s, w.Bbuf, g.fcw[2], 1, w.featb, 128, nullptr, NT, 128, 512, 0, 0, 1.f, w.bfA, w.bfB);
  fill(s, outH, 0.f, NT * 128);
  run_gat_attention(s, src, dst, E, 1, w.featb, g, 2, outH, w);
  gemm_auto(s, w.Bbuf, g.resw[2], 1, w.resb, 128, nullptr, NT, 128, 512, 0, 0, 1.f, w.bfA, w.bfB);
  ew_finish<<<(NT * 128 + 255) / 256, 256, 0, s>>>(outH, w.resb, g.bias[2], NT, 128, 0);
}

extern "C" void kernel_launch(void* const* d_in, const int* in_sizes, int n_in,
                              void* d_out, int out_size, void* d_ws, size_t ws_size,
                              hipStream_t stream) {
  const float* P = (const float*)d_in[0];
  const float* onehot[3] = {(const float*)d_in[1], (const float*)d_in[2], (const float*)d_in[3]};
  const float* ntf[3]    = {(const float*)d_in[4], (const float*)d_in[5], (const float*)d_in[6]};
  const float* featr[3]  = {(const float*)d_in[7], (const float*)d_in[8], (const float*)d_in[9]};
  const float* adj = (const float*)d_in[10];
  const int* src = (const int*)d_in[11];
  const int* dst = (const int*)d_in[12];
  int E = in_sizes[11];

  Params pp;
  build_params(P, pp);

  // workspace carve-up (bf16 regions first: keeps 16B alignment for async B128)
  WS w;
  w.bfA = (unsigned short*)d_ws;                 // up to 16M elems (onehot0 / attn)
  w.bfB = w.bfA + (size_t)16000000;              // up to 1M elems (weights / h1c / tf^T)
  float* ws = (float*)(w.bfB + (size_t)1048576);
  w.hbuf  = ws;                       ws += (size_t)NT * 256;
  w.featb = ws;                       ws += (size_t)NT * 512;
  w.Abuf  = ws;                       ws += (size_t)NT * 512;
  w.Bbuf  = ws;                       ws += (size_t)NT * 512;
  w.resb  = ws;                       ws += (size_t)NT * 128;
  w.el    = ws;                       ws += (size_t)NT * NHEADS;
  w.er    = ws;                       ws += (size_t)NT * NHEADS;
  w.mx    = ws;                       ws += (size_t)NT * NHEADS;
  w.den   = ws;                       ws += (size_t)NT * NHEADS;
  w.ebuf  = ws;                       ws += (size_t)E * NHEADS;
  w.h1c   = ws;                       ws += (size_t)4000 * 128;
  w.h2c   = ws;                       ws += (size_t)3000 * 128;
  w.attn  = ws;                       ws += (size_t)3000 * 4000;

  // output slices: emb, logits_2, h_rep, tf
  float* out    = (float*)d_out;
  float* emb    = out;
  float* logit2 = emb + (size_t)NT * HDIM;
  float* hrep   = logit2 + (size_t)NT * NCLS;
  float* tf     = hrep + (size_t)NT * HDIM;

  // -------- layer1 HGAT (one-hot inputs) --------
  const float* f1[3] = {onehot[0], onehot[1], onehot[2]};
  const int d1[3] = {4000, 3000, 2000};
  run_hgat(stream, pp.hg[0], f1, d1, ntf, src, dst, E, emb, w);

  // -------- fc_top: per-type linear into tf (type 1 replaced by completion) --------
  gemm_auto(stream, featr[0], pp.fc_top_W[0], 1, tf, 128, pp.fc_top_b[0],
            4000, 128, 128, 0, 0, 1.f, w.bfA, w.bfB);
  gemm_auto(stream, featr[2], pp.fc_top_W[2], 1, tf + (size_t)7000 * 128, 128,
            pp.fc_top_b[2], 2000, 128, 128, 0, 0, 1.f, w.bfA, w.bfB);
  fill(stream, tf + (size_t)4000 * 128, 0.f, 3000 * 128);

  // -------- attribute completion for type 1 (HGNN-AC) --------
  for (int h = 0; h < NHEADS; ++h) {
    const float* Wh = pp.hgn_W + (size_t)h * 128 * 128;
    gemm_auto(stream, emb, Wh, 1, w.h1c, 128, nullptr, 4000, 128, 128, 0, 1, 1.f,
              w.bfA, w.bfB);
    gemm_auto(stream, emb + (size_t)4000 * 128, Wh, 1, w.h2c, 128, nullptr,
              3000, 128, 128, 0, 1, 1.f, w.bfA, w.bfB);
    // logits = h2 @ h1^T : h1c is already [n][k]
    gemm_auto(stream, w.h2c, w.h1c, 0, w.attn, 4000, nullptr,
              3000, 4000, 128, 0, 0, 1.f, w.bfA, w.bfB);
    masked_softmax<<<3000, 256, 0, stream>>>(w.attn, 4000, adj, NT, 4000, 0, 4000);
    gemm_auto(stream, w.attn, tf /*feat_src rows 0..3999*/, 1,
              tf + (size_t)4000 * 128, 128, nullptr, 3000, 128, 4000,
              1 /*accumulate*/, 0, 1.f / NHEADS, w.bfA, w.bfB);
  }

  // -------- layer3 HGAT (tf parts as inputs) --------
  const float* f3[3] = {tf, tf + (size_t)4000 * 128, tf + (size_t)7000 * 128};
  const int d3[3] = {128, 128, 128};
  run_hgat(stream, pp.hg[1], f3, d3, ntf, src, dst, E, hrep, w);

  // logits_2 = h_rep @ linesW + b
  gemm_auto(stream, hrep, pp.hg[1].linW, 1, logit2, 3, pp.hg[1].linb,
            NT, 3, 128, 0, 0, 1.f, w.bfA, w.bfB);
}
